// DeepseekV3AttentionMLA_17806934409995
// MI455X (gfx1250) — matrix operand order repaired and checked by reading, not compiled, not run on hardware
//
#include <hip/hip_runtime.h>
#include <hip/hip_bf16.h>

// ---------------------------------------------------------------------------
// CDNA5 (gfx1250) wave32 WMMA helpers
// ---------------------------------------------------------------------------
typedef __attribute__((ext_vector_type(16))) __bf16 v16bf;
typedef __attribute__((ext_vector_type(8)))  float  v8f;

union ABFrag { v16bf v; unsigned int u[8]; };

__device__ inline v8f wmma_bf16(const ABFrag& a, const ABFrag& b, v8f c) {
  // D = A(16x32 bf16) * B(32x16 bf16) + C(16x16 f32)
  return __builtin_amdgcn_wmma_f32_16x16x32_bf16(
      /*neg_a=*/false, a.v, /*neg_b=*/false, b.v,
      /*c_mod=*/(short)0, c, /*reuse_a=*/false, /*reuse_b=*/false);
}

__device__ inline v8f zero8() {
  v8f z = {0.f, 0.f, 0.f, 0.f, 0.f, 0.f, 0.f, 0.f};
  return z;
}

__device__ inline unsigned short f2bf(float f) {
  unsigned int x = __float_as_uint(f);
  unsigned int r = (x + 0x7FFFu + ((x >> 16) & 1u)) >> 16;  // RNE
  return (unsigned short)r;
}

// A-fragment K index for element pair p (0..7), half = lane>>4 (contiguous: k = 16*half + 2p)
__device__ inline int a_k_of(int p, int hh) {
  return hh * 16 + 2 * p;
}

// ---------------------------------------------------------------------------
// CDNA5 async global->LDS copy (ASYNCcnt path, no VGPR round trip).
// Generic pointers to __shared__ carry the LDS offset in their low 32 bits
// (ISA 10.2 aperture rules), so the VDST operand is just the truncated addr.
// ---------------------------------------------------------------------------
__device__ inline void async_load_b128(void* lds_dst, const void* gsrc) {
  unsigned lds = (unsigned)(size_t)lds_dst;
  unsigned long long ga = (unsigned long long)gsrc;
  asm volatile("global_load_async_to_lds_b128 %0, %1, off"
               :
               : "v"(lds), "v"(ga)
               : "memory");
}

__device__ inline void wait_async0() {
  asm volatile("s_wait_asynccnt 0x0" ::: "memory");
}

// ---------------------------------------------------------------------------
// f32 -> bf16 convert
// ---------------------------------------------------------------------------
__global__ __launch_bounds__(256) void k_f32_to_bf16(
    const float* __restrict__ in, unsigned short* __restrict__ out, size_t n) {
  size_t i = (size_t)blockIdx.x * blockDim.x + threadIdx.x;
  size_t stride = (size_t)gridDim.x * blockDim.x;
  for (; i < n; i += stride) out[i] = f2bf(in[i]);
}

// ---------------------------------------------------------------------------
// bf16 GEMM: C(f32, MxN) = A(bf16, MxK row-major) * B(bf16, KxN row-major)
// Block tile 128x64, BK=32, 256 threads = 8 waves (4 x 2), wave tile 32x32.
// Double-buffered LDS, async global->LDS tile copies.
// Requires M%128==0, N%64==0, K%32==0 (true for all shapes here).
// ---------------------------------------------------------------------------
__global__ __launch_bounds__(256) void k_gemm_bf16(
    const unsigned short* __restrict__ A, const unsigned short* __restrict__ B,
    float* __restrict__ C, int M, int N, int K) {
  __shared__ unsigned short As[2][128 * 32];
  __shared__ unsigned short Bs[2][32 * 64];

  const int tid  = threadIdx.x;
  const int wid  = tid >> 5;
  const int lane = tid & 31;
  const int ln15 = lane & 15;
  const int hh   = lane >> 4;
  const int waveM = wid >> 1;  // 0..3 (32 rows each)
  const int waveN = wid & 1;   // 0..1 (32 cols each)
  const int rBlock = blockIdx.y * 128;
  const int cBlock = blockIdx.x * 64;

  // per-thread tile-chunk coordinates
  const int aRow0 = tid >> 2,        aCol0 = (tid & 3) * 8;        // chunk tid
  const int aRow1 = (tid + 256) >> 2, aCol1 = (tid & 3) * 8;       // chunk tid+256
  const int bRow  = tid >> 3,        bCol  = (tid & 7) * 8;

  auto issue_tile = [&](int kt, int buf) {
    const int kBase = kt * 32;
    async_load_b128(As[buf] + aRow0 * 32 + aCol0,
                    A + (size_t)(rBlock + aRow0) * K + kBase + aCol0);
    async_load_b128(As[buf] + aRow1 * 32 + aCol1,
                    A + (size_t)(rBlock + aRow1) * K + kBase + aCol1);
    async_load_b128(Bs[buf] + bRow * 64 + bCol,
                    B + (size_t)(kBase + bRow) * N + cBlock + bCol);
  };

  v8f acc[2][2];
  for (int i = 0; i < 2; ++i)
    for (int j = 0; j < 2; ++j) acc[i][j] = zero8();

  const int nk = K >> 5;
  issue_tile(0, 0);
  wait_async0();
  __syncthreads();

  for (int kt = 0; kt < nk; ++kt) {
    const int buf = kt & 1;
    if (kt + 1 < nk) {
      issue_tile(kt + 1, buf ^ 1);  // DMA overlaps the WMMAs below
      __builtin_prefetch(A + (size_t)(rBlock + aRow0) * K + (kt + 2) * 32, 0, 1);
    }

    ABFrag afr[2], bfr[2];
#pragma unroll
    for (int i = 0; i < 2; ++i) {
      int r0 = waveM * 32 + i * 16 + ln15;
#pragma unroll
      for (int p = 0; p < 8; ++p)
        afr[i].u[p] = *(const unsigned int*)(As[buf] + r0 * 32 + a_k_of(p, hh));
    }
#pragma unroll
    for (int j = 0; j < 2; ++j) {
      int c0 = waveN * 32 + j * 16 + ln15;
#pragma unroll
      for (int p = 0; p < 8; ++p) {
        int k = hh * 16 + 2 * p;
        unsigned int lo = Bs[buf][k * 64 + c0];
        unsigned int hi = Bs[buf][(k + 1) * 64 + c0];
        bfr[j].u[p] = lo | (hi << 16);
      }
    }
#pragma unroll
    for (int i = 0; i < 2; ++i)
#pragma unroll
      for (int j = 0; j < 2; ++j) acc[i][j] = wmma_bf16(afr[i], bfr[j], acc[i][j]);

    wait_async0();
    __syncthreads();
  }

  // C layout: element e -> row e + 8*half, col = lane&15
#pragma unroll
  for (int i = 0; i < 2; ++i)
#pragma unroll
    for (int j = 0; j < 2; ++j)
#pragma unroll
      for (int e = 0; e < 8; ++e) {
        int row = rBlock + waveM * 32 + i * 16 + e + 8 * hh;
        int col = cBlock + waveN * 32 + j * 16 + ln15;
        C[(size_t)row * N + col] = acc[i][j][e];
      }
}

// ---------------------------------------------------------------------------
// RMSNorm (f32 in, bf16 out), one block per row
// ---------------------------------------------------------------------------
__global__ __launch_bounds__(256) void k_rmsnorm_bf16(
    const float* __restrict__ in, int ld_in, const float* __restrict__ w,
    unsigned short* __restrict__ out, int ld_out, int C) {
  __shared__ float red[8];
  __shared__ float stot;
  const int row = blockIdx.x;
  const float* x = in + (size_t)row * ld_in;
  float ss = 0.f;
  for (int i = threadIdx.x; i < C; i += 256) {
    float v = x[i];
    ss += v * v;
  }
  for (int off = 16; off > 0; off >>= 1) ss += __shfl_xor(ss, off, 32);
  if ((threadIdx.x & 31) == 0) red[threadIdx.x >> 5] = ss;
  __syncthreads();
  if (threadIdx.x == 0) {
    float t = 0.f;
    for (int i = 0; i < 8; ++i) t += red[i];
    stot = t;
  }
  __syncthreads();
  const float rinv = rsqrtf(stot / (float)C + 1e-6f);
  unsigned short* o = out + (size_t)row * ld_out;
  for (int i = threadIdx.x; i < C; i += 256) o[i] = f2bf(x[i] * rinv * w[i]);
}

// ---------------------------------------------------------------------------
// YaRN RoPE + pack Q/K/V to bf16. One block per token t.
//  qf:  (T, 32, 192) f32 ; latent: (T, 576) f32 (cols 512.. = k_pe)
//  kvf: (T, 32, 256) f32
//  Qb/Kb: (T, 32, 192) bf16, Vb: (T, 32, 128) bf16. Scaling folded into Q.
// ---------------------------------------------------------------------------
__global__ __launch_bounds__(256) void k_rope_pack(
    const int* __restrict__ pos, const float* __restrict__ qf,
    const float* __restrict__ latent, const float* __restrict__ kvf,
    unsigned short* __restrict__ Qb, unsigned short* __restrict__ Kb,
    unsigned short* __restrict__ Vb) {
  __shared__ float cs[32], sn[32], kr[64];
  const int t = blockIdx.x;
  const float p = (float)pos[t];
  if (threadIdx.x < 32) {
    const int j = threadIdx.x;
    const float TWO_PI = 6.28318530717958647692f;
    float inv_extra = __powf(10000.f, -(float)j / 32.f);
    float inv_inter = inv_extra / 40.f;
    float denom = 2.f * logf(10000.f);
    float low  = fmaxf(floorf(64.f * logf(4096.f / (32.f * TWO_PI)) / denom), 0.f);
    float high = fminf(ceilf(64.f * logf(4096.f / (1.f * TWO_PI)) / denom), 63.f);
    float ramp = ((float)j - low) / fmaxf(high - low, 0.001f);
    ramp = fminf(fmaxf(ramp, 0.f), 1.f);
    float mask = 1.f - ramp;
    float invf = inv_inter * (1.f - mask) + inv_extra * mask;
    float fr = p * invf;
    cs[j] = __cosf(fr);
    sn[j] = __sinf(fr);
  }
  __syncthreads();
  if (threadIdx.x < 32) {
    const int j = threadIdx.x;
    float x1 = latent[(size_t)t * 576 + 512 + 2 * j];
    float x2 = latent[(size_t)t * 576 + 512 + 2 * j + 1];
    kr[2 * j]     = x1 * cs[j] - x2 * sn[j];
    kr[2 * j + 1] = x2 * cs[j] + x1 * sn[j];
  }
  __syncthreads();

  const float mscale  = 0.1f * logf(40.f) + 1.f;
  const float scaling = rsqrtf(192.f) * mscale * mscale;
  const float* qrow  = qf  + (size_t)t * 6144;
  const float* kvrow = kvf + (size_t)t * 8192;
  unsigned short* Qo = Qb + (size_t)t * 6144;
  unsigned short* Ko = Kb + (size_t)t * 6144;
  unsigned short* Vo = Vb + (size_t)t * 4096;

  for (int idx = threadIdx.x; idx < 6144; idx += 256) {
    int h = idx / 192, d = idx - h * 192;
    float qv;
    if (d < 128) {
      qv = qrow[idx];
    } else {
      int j = (d - 128) >> 1;
      float x1 = qrow[h * 192 + 128 + 2 * j];
      float x2 = qrow[h * 192 + 128 + 2 * j + 1];
      qv = ((d - 128) & 1) ? (x2 * cs[j] + x1 * sn[j]) : (x1 * cs[j] - x2 * sn[j]);
    }
    Qo[idx] = f2bf(qv * scaling);
    float kv = (d < 128) ? kvrow[h * 256 + d] : kr[d - 128];
    Ko[idx] = f2bf(kv);
  }
  for (int idx = threadIdx.x; idx < 4096; idx += 256) {
    int h = idx >> 7, dv = idx & 127;
    Vo[idx] = f2bf(kvrow[h * 256 + 128 + dv]);
  }
}

// ---------------------------------------------------------------------------
// Causal flash attention per head. Block = 128 threads = 4 waves, each wave
// owns 16 query rows (64 q rows / block). Key blocks of 32 staged in LDS,
// double-buffered with async global->LDS copies.
// grid = (T/64, H). Output: attn bf16 (T, H*128).
// ---------------------------------------------------------------------------
__global__ __launch_bounds__(128) void k_mla_attn(
    const unsigned short* __restrict__ Qb, const unsigned short* __restrict__ Kb,
    const unsigned short* __restrict__ Vb, unsigned short* __restrict__ Ob) {
  __shared__ unsigned short Kt[2][32 * 192];
  __shared__ unsigned short Vt[2][32 * 128];
  __shared__ unsigned short Ps[4][16 * 32];

  const int h      = blockIdx.y;
  const int qBlock = blockIdx.x;
  const int tid    = threadIdx.x;
  const int wid    = tid >> 5;
  const int lane   = tid & 31;
  const int ln15   = lane & 15;
  const int hh     = lane >> 4;
  const int qBase  = qBlock * 64 + wid * 16;

  auto issue_kv = [&](int kb, int buf) {
    const int kKey = kb * 32;
#pragma unroll
    for (int c = 0; c < 6; ++c) {
      int chunk = tid + c * 128;
      int row = chunk / 24;
      int col8 = (chunk % 24) * 8;
      async_load_b128(Kt[buf] + row * 192 + col8,
                      Kb + ((size_t)(kKey + row) * 32 + h) * 192 + col8);
    }
#pragma unroll
    for (int c = 0; c < 4; ++c) {
      int chunk = tid + c * 128;
      int row = chunk >> 4;
      int col8 = (chunk & 15) * 8;
      async_load_b128(Vt[buf] + row * 128 + col8,
                      Vb + ((size_t)(kKey + row) * 32 + h) * 128 + col8);
    }
  };

  // Q fragments: 6 chunks of K=32 covering D_QK=192; A-frag in contiguous k order (k = 16*half + 2p)
  ABFrag qfr[6];
  {
    const unsigned short* qp = Qb + ((size_t)(qBase + ln15) * 32 + h) * 192;
#pragma unroll
    for (int c = 0; c < 6; ++c)
#pragma unroll
      for (int p = 0; p < 8; ++p)
        qfr[c].u[p] = *(const unsigned int*)(qp + c * 32 + a_k_of(p, hh));
  }

  v8f o[8];
#pragma unroll
  for (int j = 0; j < 8; ++j) o[j] = zero8();
  float mrow[8], lrow[8];
#pragma unroll
  for (int e = 0; e < 8; ++e) {
    mrow[e] = -3.0e38f;
    lrow[e] = 0.f;
  }

  const int nkb = qBlock * 2 + 2;  // key blocks of 32 covering [0, qBase_max]
  issue_kv(0, 0);
  wait_async0();
  __syncthreads();

  for (int kb = 0; kb < nkb; ++kb) {
    const int buf = kb & 1;
    const int kKey = kb * 32;
    if (kb + 1 < nkb) issue_kv(kb + 1, buf ^ 1);  // DMA overlaps compute

    // Scores: two 16x16 tiles (keys kKey..+15, kKey+16..+31)
    v8f s[2];
#pragma unroll
    for (int nt = 0; nt < 2; ++nt) {
      v8f accs = zero8();
      const int key = nt * 16 + ln15;  // B-frag: lane holds column (key)
#pragma unroll
      for (int c = 0; c < 6; ++c) {
        ABFrag bfr;
#pragma unroll
        for (int p = 0; p < 8; ++p) {
          int k = c * 32 + hh * 16 + 2 * p;
          bfr.u[p] = *(const unsigned int*)(Kt[buf] + key * 192 + k);
        }
        accs = wmma_bf16(qfr[c], bfr, accs);
      }
      s[nt] = accs;
    }

    // Causal mask + online softmax (rows in elements, cols in lanes)
#pragma unroll
    for (int e = 0; e < 8; ++e) {
      const int qrow = qBase + e + 8 * hh;
      float m2 = mrow[e];
#pragma unroll
      for (int nt = 0; nt < 2; ++nt) {
        int key = kKey + nt * 16 + ln15;
        float v = (key <= qrow) ? s[nt][e] : -3.0e38f;
        s[nt][e] = v;
        m2 = fmaxf(m2, v);
      }
      for (int off = 1; off < 16; off <<= 1) m2 = fmaxf(m2, __shfl_xor(m2, off, 32));
      const float alpha = __expf(mrow[e] - m2);
      float rs = 0.f;
#pragma unroll
      for (int nt = 0; nt < 2; ++nt) {
        float pv = __expf(s[nt][e] - m2);
        s[nt][e] = pv;
        rs += pv;
      }
      for (int off = 1; off < 16; off <<= 1) rs += __shfl_xor(rs, off, 32);
      lrow[e] = lrow[e] * alpha + rs;
      mrow[e] = m2;
#pragma unroll
      for (int j = 0; j < 8; ++j) o[j][e] *= alpha;
    }

    // P (C-layout) -> per-wave LDS slab -> A-layout fragment
    unsigned short* ps = Ps[wid];
#pragma unroll
    for (int e = 0; e < 8; ++e) {
      int prow = e + 8 * hh;
      ps[prow * 32 + ln15]      = f2bf(s[0][e]);
      ps[prow * 32 + 16 + ln15] = f2bf(s[1][e]);
    }
    __syncthreads();

    ABFrag pfr;
#pragma unroll
    for (int p = 0; p < 8; ++p)
      pfr.u[p] = *(const unsigned int*)(ps + ln15 * 32 + a_k_of(p, hh));

    // O += P(16x32) * V(32x128)
#pragma unroll
    for (int j = 0; j < 8; ++j) {
      ABFrag vfr;
      const int col = j * 16 + ln15;
#pragma unroll
      for (int p = 0; p < 8; ++p) {
        int d0 = hh * 16 + 2 * p;
        unsigned int lo = Vt[buf][d0 * 128 + col];
        unsigned int hi = Vt[buf][(d0 + 1) * 128 + col];
        vfr.u[p] = lo | (hi << 16);
      }
      o[j] = wmma_bf16(pfr, vfr, o[j]);
    }

    wait_async0();
    __syncthreads();
  }

  // Normalize and store bf16 attn (T, H*128)
#pragma unroll
  for (int j = 0; j < 8; ++j)
#pragma unroll
    for (int e = 0; e < 8; ++e) {
      int qrow = qBase + e + 8 * hh;
      float val = o[j][e] / lrow[e];
      Ob[(size_t)qrow * 4096 + h * 128 + j * 16 + ln15] = f2bf(val);
    }
}

// ---------------------------------------------------------------------------
// Host launcher
// ---------------------------------------------------------------------------
static inline int cvt_blocks(size_t n) {
  size_t b = (n + 255) / 256;
  return (int)(b > 4096 ? 4096 : b);
}

extern "C" void kernel_launch(void* const* d_in, const int* in_sizes, int n_in,
                              void* d_out, int out_size, void* d_ws, size_t ws_size,
                              hipStream_t stream) {
  (void)in_sizes; (void)n_in; (void)out_size; (void)ws_size;
  const int*   positions = (const int*)d_in[0];
  const float* hidden    = (const float*)d_in[1];
  const float* Wqa       = (const float*)d_in[2];
  const float* q_a_ln    = (const float*)d_in[3];
  const float* Wqb       = (const float*)d_in[4];
  const float* Wkva      = (const float*)d_in[5];
  const float* kv_a_ln   = (const float*)d_in[6];
  const float* Wkvb      = (const float*)d_in[7];
  const float* Wo        = (const float*)d_in[8];
  float* out = (float*)d_out;

  const int T = 2048, HID = 4096;

  // workspace carve (256B aligned)
  char* p = (char*)d_ws;
  auto alloc = [&](size_t bytes) -> char* {
    char* r = p;
    p += (bytes + 255) & ~(size_t)255;
    return r;
  };
  unsigned short* hid_b  = (unsigned short*)alloc((size_t)T * HID * 2);
  unsigned short* Wqa_b  = (unsigned short*)alloc((size_t)HID * 1536 * 2);
  unsigned short* Wqb_b  = (unsigned short*)alloc((size_t)1536 * 6144 * 2);
  unsigned short* Wkva_b = (unsigned short*)alloc((size_t)HID * 576 * 2);
  unsigned short* Wkvb_b = (unsigned short*)alloc((size_t)512 * 8192 * 2);
  unsigned short* Wo_b   = (unsigned short*)alloc((size_t)HID * HID * 2);
  float*          qa_f   = (float*)alloc((size_t)T * 1536 * 4);
  unsigned short* qan_b  = (unsigned short*)alloc((size_t)T * 1536 * 2);
  float*          lat_f  = (float*)alloc((size_t)T * 576 * 4);
  unsigned short* kvan_b = (unsigned short*)alloc((size_t)T * 512 * 2);
  float*          q_f    = (float*)alloc((size_t)T * 6144 * 4);
  float*          kv_f   = (float*)alloc((size_t)T * 8192 * 4);
  unsigned short* Qb     = (unsigned short*)alloc((size_t)T * 6144 * 2);
  unsigned short* Kb     = (unsigned short*)alloc((size_t)T * 6144 * 2);
  unsigned short* Vb     = (unsigned short*)alloc((size_t)T * 4096 * 2);
  unsigned short* attn_b = (unsigned short*)alloc((size_t)T * 4096 * 2);

  // 1) bf16 conversions
  {
    size_t n;
    n = (size_t)T * HID;     k_f32_to_bf16<<<cvt_blocks(n), 256, 0, stream>>>(hidden, hid_b, n);
    n = (size_t)HID * 1536;  k_f32_to_bf16<<<cvt_blocks(n), 256, 0, stream>>>(Wqa, Wqa_b, n);
    n = (size_t)1536 * 6144; k_f32_to_bf16<<<cvt_blocks(n), 256, 0, stream>>>(Wqb, Wqb_b, n);
    n = (size_t)HID * 576;   k_f32_to_bf16<<<cvt_blocks(n), 256, 0, stream>>>(Wkva, Wkva_b, n);
    n = (size_t)512 * 8192;  k_f32_to_bf16<<<cvt_blocks(n), 256, 0, stream>>>(Wkvb, Wkvb_b, n);
    n = (size_t)HID * HID;   k_f32_to_bf16<<<cvt_blocks(n), 256, 0, stream>>>(Wo, Wo_b, n);
  }

  // 2) q_a = hidden @ Wqa   (2048 x 1536, K=4096)
  k_gemm_bf16<<<dim3(1536 / 64, T / 128), 256, 0, stream>>>(hid_b, Wqa_b, qa_f, T, 1536, HID);
  // 3) latent = hidden @ Wkva (2048 x 576, K=4096)
  k_gemm_bf16<<<dim3(576 / 64, T / 128), 256, 0, stream>>>(hid_b, Wkva_b, lat_f, T, 576, HID);

  // 4) RMSNorms -> bf16
  k_rmsnorm_bf16<<<T, 256, 0, stream>>>(qa_f, 1536, q_a_ln, qan_b, 1536, 1536);
  k_rmsnorm_bf16<<<T, 256, 0, stream>>>(lat_f, 576, kv_a_ln, kvan_b, 512, 512);

  // 5) q = q_a_norm @ Wqb   (2048 x 6144, K=1536)
  k_gemm_bf16<<<dim3(6144 / 64, T / 128), 256, 0, stream>>>(qan_b, Wqb_b, q_f, T, 6144, 1536);
  // 6) kv = kv_a_norm @ Wkvb (2048 x 8192, K=512)
  k_gemm_bf16<<<dim3(8192 / 64, T / 128), 256, 0, stream>>>(kvan_b, Wkvb_b, kv_f, T, 8192, 512);

  // 7) RoPE + pack Q/K/V bf16 (scaling folded into Q)
  k_rope_pack<<<T, 256, 0, stream>>>(positions, q_f, lat_f, kv_f, Qb, Kb, Vb);

  // 8) causal flash attention per head -> attn bf16 (T, 4096)
  k_mla_attn<<<dim3(T / 64, 32), 128, 0, stream>>>(Qb, Kb, Vb, attn_b);

  // 9) out = attn @ Wo (2048 x 4096, K=4096), f32 straight to d_out
  k_gemm_bf16<<<dim3(HID / 64, T / 128), 256, 0, stream>>>(attn_b, Wo_b, out, T, HID, HID);
}
